// gcompool_62792421868057
// MI455X (gfx1250) — compile-verified, matrix operand order checked by reading, not run on hardware
//
#include <hip/hip_runtime.h>
#include <hip/hip_bf16.h>

typedef __attribute__((ext_vector_type(16))) _Float16 v16h;
typedef __attribute__((ext_vector_type(8)))  _Float16 v8h;
typedef __attribute__((ext_vector_type(8)))  float    v8f;
typedef __attribute__((ext_vector_type(4)))  float    v4f;

#define GS     2048
#define PARAM  64
#define OGS    1024
#define FDIM   128   // PARAM * C
#define PARAMO 64
#define NTHREADS 256

// One workgroup per batch row:
//  Phase 1: bitonic argsort (descending by x[b,g,63], ties -> lower index) in LDS,
//           plus stage trafo into LDS as f16, transposed (col-major: s_trafT[col*128 + f]).
//  Phase 2: per-wave WMMA f16 GEMM  xs(1024x128) @ trafo(128x64), f32 accumulate.
__global__ __launch_bounds__(NTHREADS) void gcompool_62792421868057_kernel(
    const float* __restrict__ x, const float* __restrict__ trafo, float* __restrict__ out)
{
    __shared__ float    s_key[GS];
    __shared__ unsigned s_idx[GS];
    __shared__ _Float16 s_trafT[PARAMO * FDIM];   // [col][f], 16KB

    const int tid = threadIdx.x;
    const int b   = blockIdx.x;
    const float* xb = x + (size_t)b * GS * PARAM;

    // ---- Phase 1a: load keys + indices, stage trafo (f32 -> f16, transposed) ----
    for (int g = tid; g < GS; g += NTHREADS) {
        s_key[g] = xb[(size_t)g * PARAM + (PARAM - 1)];
        s_idx[g] = (unsigned)g;
    }
    for (int i = tid; i < FDIM * PARAMO; i += NTHREADS) {
        int f = i >> 6;        // 0..127
        int c = i & 63;        // 0..63
        s_trafT[c * FDIM + f] = (_Float16)trafo[i];
    }
    __syncthreads();

    // ---- Phase 1b: bitonic sort, descending by key, ascending index on ties ----
    for (int k = 2; k <= GS; k <<= 1) {
        for (int j = k >> 1; j > 0; j >>= 1) {
            #pragma unroll
            for (int t = 0; t < GS / NTHREADS; ++t) {
                int i   = tid + t * NTHREADS;
                int ixj = i ^ j;
                if (ixj > i) {
                    float    ka = s_key[i],  kb = s_key[ixj];
                    unsigned ia = s_idx[i],  ib = s_idx[ixj];
                    // "a before b" under the desired (descending, stable) order
                    bool aFirst = (ka > kb) || (ka == kb && ia < ib);
                    bool up     = ((i & k) == 0);       // up-blocks hold the desired order
                    bool doSwap = up ? !aFirst : aFirst;
                    if (doSwap) {
                        s_key[i] = kb; s_key[ixj] = ka;
                        s_idx[i] = ib; s_idx[ixj] = ia;
                    }
                }
            }
            __syncthreads();
        }
    }
    // s_idx[] now holds the descending-sort permutation ("order").

    // ---- Phase 2: WMMA GEMM ----
    const int lane  = tid & 31;
    const int wv    = tid >> 5;        // wave 0..7
    const int l     = lane & 15;       // column-in-tile / row-in-tile lane index
    const int hi    = lane >> 4;       // half-wave select
    const int kbase = hi * 8;          // K offset per 16-bit WMMA operand layout

    // B fragments: bf[kc][n] covers K = kc*32..kc*32+31, N = n*16..n*16+15.
    // Lane l (+ half) holds col N=n*16+l, k = kbase+{0..7} and kbase+16+{0..7}.
    v16h bf[4][4];
    #pragma unroll
    for (int kc = 0; kc < 4; ++kc) {
        #pragma unroll
        for (int n = 0; n < 4; ++n) {
            const _Float16* p = &s_trafT[(n * 16 + l) * FDIM + kc * 32 + kbase];
            v8h lo8 = *(const v8h*)p;          // k = kbase + 0..7
            v8h hi8 = *(const v8h*)(p + 16);   // k = kbase + 16..23
            v16h f;
            #pragma unroll
            for (int i = 0; i < 8; ++i) { f[i] = lo8[i]; f[i + 8] = hi8[i]; }
            bf[kc][n] = f;
        }
    }

    // Each wave: 8 row-tiles of 16 output rows.
    #pragma unroll 1
    for (int rr = 0; rr < 8; ++rr) {
        const int r = wv * 8 + rr;            // row-tile 0..63
        const int o = r * 16 + l;             // output row handled by this lane's M slot

        v8f acc[4] = {};
        #pragma unroll
        for (int kc = 0; kc < 4; ++kc) {
            // feature f = kc*32 + k  ->  source row order[2o + (f>=64)], col f&63
            unsigned srcRow = s_idx[2 * o + (kc >> 1)];
            const float* src = xb + (size_t)srcRow * PARAM + (kc & 1) * 32 + kbase;
            v4f q0 = ((const v4f*)src)[0];          // k = kbase + 0..3
            v4f q1 = ((const v4f*)src)[1];          // k = kbase + 4..7
            v4f q2 = ((const v4f*)(src + 16))[0];   // k = kbase + 16..19
            v4f q3 = ((const v4f*)(src + 16))[1];   // k = kbase + 20..23
            v16h a;
            #pragma unroll
            for (int i = 0; i < 4; ++i) {
                a[i]      = (_Float16)q0[i];
                a[4 + i]  = (_Float16)q1[i];
                a[8 + i]  = (_Float16)q2[i];
                a[12 + i] = (_Float16)q3[i];
            }
            #pragma unroll
            for (int n = 0; n < 4; ++n) {
                acc[n] = __builtin_amdgcn_wmma_f32_16x16x32_f16(
                    false, a, false, bf[kc][n], (short)0, acc[n], false, false);
            }
        }

        // C/D layout: lane 0-15 -> M = v, lane 16-31 -> M = v+8; N = lane&15.
        float* ob = out + ((size_t)b * OGS + (size_t)r * 16 + hi * 8) * PARAMO;
        #pragma unroll
        for (int n = 0; n < 4; ++n) {
            #pragma unroll
            for (int v = 0; v < 8; ++v) {
                ob[(size_t)v * PARAMO + n * 16 + l] = acc[n][v];
            }
        }
    }
}

extern "C" void kernel_launch(void* const* d_in, const int* in_sizes, int n_in,
                              void* d_out, int out_size, void* d_ws, size_t ws_size,
                              hipStream_t stream) {
    const float* x     = (const float*)d_in[0];   // (B, 2048, 64) f32
    const float* trafo = (const float*)d_in[1];   // (128, 64) f32
    float* out         = (float*)d_out;           // (B, 1024, 64) f32
    (void)d_ws; (void)ws_size; (void)n_in;

    const int B = in_sizes[0] / (GS * PARAM);     // 512
    gcompool_62792421868057_kernel<<<B, NTHREADS, 0, stream>>>(x, trafo, out);
}
